// DiTACEBlock_58411555226344
// MI455X (gfx1250) — matrix operand order repaired and checked by reading, not compiled, not run on hardware
//
#include <hip/hip_runtime.h>
#include <hip/hip_bf16.h>

// ---------------------------------------------------------------------------
// Types for CDNA5 WMMA fragments (wave32, 16x16x32 f16 -> f32)
// ---------------------------------------------------------------------------
typedef __attribute__((ext_vector_type(16))) _Float16 v16h;
typedef __attribute__((ext_vector_type(8)))  float    v8f;
typedef __attribute__((ext_vector_type(4)))  unsigned int u32x4;

union FragH { v16h h; u32x4 q[2]; };

// ---------------------------------------------------------------------------
// gfx1250 async global->LDS copy path (ASYNCcnt), with safe fallback
// ---------------------------------------------------------------------------
#if defined(__has_builtin)
#if __has_builtin(__builtin_amdgcn_global_load_async_to_lds_b128)
#define HAVE_ASYNC_COPY 1
#endif
#endif
#ifndef HAVE_ASYNC_COPY
#define HAVE_ASYNC_COPY 0
#endif

// Builtin signature (from compiler diagnostic): param0 is
//   int __vector(4) addrspace(1)*  (printed as "... int __device__ *")
// so: (v4i AS1* gsrc, v4i AS3* ldst, Ii offset, Ii cpol)
typedef int v4i __attribute__((vector_size(16)));
typedef __attribute__((address_space(1))) v4i gv4i;
typedef __attribute__((address_space(3))) v4i lv4i;

__device__ inline void async_cp_b128(const _Float16* g, _Float16* l) {
#if HAVE_ASYNC_COPY
  __builtin_amdgcn_global_load_async_to_lds_b128((gv4i*)g, (lv4i*)l, 0, 0);
#else
  *(u32x4*)l = *(const u32x4*)g;
#endif
}

template <int N>
__device__ inline void wait_async() {
#if HAVE_ASYNC_COPY
#if __has_builtin(__builtin_amdgcn_s_wait_asynccnt)
  __builtin_amdgcn_s_wait_asynccnt(N);
#else
  asm volatile("s_wait_asynccnt %0" ::"i"(N) : "memory");
#endif
#endif
}

__device__ inline v8f wmma_f16(const FragH& a, const FragH& b, v8f c) {
  // D = A(16x32) x B(32x16) + C, f32 accumulate
  return __builtin_amdgcn_wmma_f32_16x16x32_f16(false, a.h, false, b.h,
                                                (short)0, c, false, false);
}

// A fragment (16x32 f16): lane L holds row L%16.
// lanes 0-15 : K = 0..7   (q[0]) and 16..23 (q[1])
// lanes 16-31: K = 8..15  (q[0]) and 24..31 (q[1])
__device__ inline void load_frag_a(FragH& f, const _Float16* p0, int ld,
                                   int rowBase, int lane) {
  const int lr = lane & 15, hi = lane >> 4;
  const _Float16* p = p0 + (size_t)(rowBase + lr) * ld;
  f.q[0] = *(const u32x4*)(p + hi * 8);
  f.q[1] = *(const u32x4*)(p + 16 + hi * 8);
}

// B fragment (32x16 f16) where lds holds B^T rows (n-major, K contiguous):
// lane L holds column n = L%16, K = 8*(L>=16) + 0..7 (q[0]) and 16 + 8*(L>=16) + 0..7 (q[1]).
__device__ inline void load_frag_b(FragH& f, const _Float16* p0, int ld,
                                   int nBase, int kOff, int lane) {
  const int n = lane & 15, hi = lane >> 4;
  const _Float16* p = p0 + (size_t)(nBase + n) * ld + kOff + hi * 8;
  f.q[0] = *(const u32x4*)p;
  f.q[1] = *(const u32x4*)(p + 16);
}

// ---------------------------------------------------------------------------
// Elementwise helpers
// ---------------------------------------------------------------------------
__global__ __launch_bounds__(256) void f32_to_f16_kernel(
    const float* __restrict__ in, _Float16* __restrict__ out, int n) {
  int i = (blockIdx.x * 256 + threadIdx.x) * 4;
  if (i + 3 < n) {
    out[i + 0] = (_Float16)in[i + 0];
    out[i + 1] = (_Float16)in[i + 1];
    out[i + 2] = (_Float16)in[i + 2];
    out[i + 3] = (_Float16)in[i + 3];
  } else {
    for (int j = i; j < n; ++j) out[j] = (_Float16)in[j];
  }
}

// LayerNorm over C=2048 + adaLN modulation: out = ln(x)*(1+scale)+shift (f16)
__global__ __launch_bounds__(256) void ln_mod_kernel(
    const float* __restrict__ x, const float* __restrict__ t6,
    const float* __restrict__ sst, _Float16* __restrict__ out,
    int C, int scaleIdx, int shiftIdx) {
  const int row = blockIdx.x, tid = threadIdx.x;
  const float* xr = x + (size_t)row * C;
  float vals[8];
  float s = 0.f, ss = 0.f;
  #pragma unroll
  for (int i = 0; i < 8; ++i) {
    float v = xr[tid + i * 256];
    vals[i] = v; s += v; ss += v * v;
  }
  #pragma unroll
  for (int off = 16; off; off >>= 1) {
    s  += __shfl_xor(s, off, 32);
    ss += __shfl_xor(ss, off, 32);
  }
  __shared__ float rs[8], rss[8];
  const int w = tid >> 5;
  if ((tid & 31) == 0) { rs[w] = s; rss[w] = ss; }
  __syncthreads();
  s = 0.f; ss = 0.f;
  #pragma unroll
  for (int j = 0; j < 8; ++j) { s += rs[j]; ss += rss[j]; }
  const float mean = s / (float)C;
  const float var  = ss / (float)C - mean * mean;
  const float inv  = rsqrtf(var + 1e-6f);
  const float* tr = t6 + (size_t)row * 6 * C;
  #pragma unroll
  for (int i = 0; i < 8; ++i) {
    int c = tid + i * 256;
    float scale = 1.f + sst[scaleIdx * C + c] + tr[scaleIdx * C + c];
    float shift =       sst[shiftIdx * C + c] + tr[shiftIdx * C + c];
    out[(size_t)row * C + c] = (_Float16)((vals[i] - mean) * inv * scale + shift);
  }
}

// RoPE in-place on f16 (B,S,H,Dh): one thread per (row, head, pair)
__global__ __launch_bounds__(256) void rope_kernel(
    _Float16* __restrict__ qk, const float* __restrict__ freqs,
    int total, int H, int Dh2, int S) {
  int i = blockIdx.x * 256 + threadIdx.x;
  if (i >= total) return;
  const int dp = i % Dh2;
  const int hh = (i / Dh2) % H;
  const int row = i / (Dh2 * H);
  const int s = row % S;
  const float f = freqs[s * Dh2 + dp];
  const float c = __cosf(f), sn = __sinf(f);
  const size_t base = ((size_t)row * H + hh) * (size_t)(2 * Dh2) + 2 * dp;
  const float a = (float)qk[base], b = (float)qk[base + 1];
  qk[base]     = (_Float16)(a * c - b * sn);
  qk[base + 1] = (_Float16)(a * sn + b * c);
}

// ---------------------------------------------------------------------------
// Tiled WMMA GEMM:  out[M,N] = A[M,K] @ W[N,K]^T + bias, fused epilogues.
// Block: 128x128 tile, 256 threads (8 waves, 4m x 2n), wave = 32x64 (2x4 WMMA).
// Double-buffered LDS staging via async global->LDS copies (ASYNCcnt).
// ---------------------------------------------------------------------------
enum { EPI_H = 0, EPI_GELU = 1, EPI_RES = 2, EPI_GATERES = 3 };

template <int EPI>
__global__ __launch_bounds__(256) void gemm_wmma_kernel(
    const _Float16* __restrict__ A, const _Float16* __restrict__ W,
    const float* __restrict__ bias,
    _Float16* __restrict__ outH, float* __restrict__ outF,
    const float* __restrict__ resid, const float* __restrict__ t6,
    const float* __restrict__ sst, int gateIdx,
    int M, int N, int K) {
  __shared__ __align__(16) _Float16 lA[2][128 * 32];
  __shared__ __align__(16) _Float16 lB[2][128 * 32];
  const int tid = threadIdx.x;
  const int lane = tid & 31, w = tid >> 5;
  const int wm = (w & 3) * 32;   // 4 waves along M (32 rows each)
  const int wn = (w >> 2) * 64;  // 2 waves along N (64 cols each)
  const int bm = blockIdx.y * 128, bn = blockIdx.x * 128;

  v8f acc[2][4];
  #pragma unroll
  for (int mi = 0; mi < 2; ++mi)
    #pragma unroll
    for (int ni = 0; ni < 4; ++ni)
      #pragma unroll
      for (int r = 0; r < 8; ++r) acc[mi][ni][r] = 0.f;

  // Tile staging: both A (128x32) and B (128x32) use the same thread map:
  // row = tid/2, 16 halves per thread (two b128 ops).
  const int tr = tid >> 1, seg = (tid & 1) * 16;
  const _Float16* gA = A + (size_t)(bm + tr) * K + seg;
  const _Float16* gB = W + (size_t)(bn + tr) * K + seg;

  auto stage = [&](int sel, int kb) {
    _Float16* dA = &lA[sel][tr * 32 + seg];
    _Float16* dB = &lB[sel][tr * 32 + seg];
    async_cp_b128(gA + kb,     dA);
    async_cp_b128(gA + kb + 8, dA + 8);
    async_cp_b128(gB + kb,     dB);
    async_cp_b128(gB + kb + 8, dB + 8);
  };

  stage(0, 0);
  int sel = 0;
  for (int kb = 0; kb < K; kb += 32) {
    if (kb + 32 < K) {
      stage(sel ^ 1, kb + 32);  // prefetch next tile into spare buffer
      wait_async<4>();          // current tile's 4 async ops have landed
    } else {
      wait_async<0>();
    }
    __syncthreads();

    FragH fa0, fa1;
    load_frag_a(fa0, lA[sel], 32, wm,      lane);
    load_frag_a(fa1, lA[sel], 32, wm + 16, lane);
    #pragma unroll
    for (int ni = 0; ni < 4; ++ni) {
      FragH fb;
      load_frag_b(fb, lB[sel], 32, wn + ni * 16, 0, lane);
      acc[0][ni] = wmma_f16(fa0, fb, acc[0][ni]);
      acc[1][ni] = wmma_f16(fa1, fb, acc[1][ni]);
    }
    __syncthreads();
    sel ^= 1;
  }

  const int hi = lane >> 4, cl = lane & 15;
  #pragma unroll
  for (int mi = 0; mi < 2; ++mi)
    #pragma unroll
    for (int ni = 0; ni < 4; ++ni)
      #pragma unroll
      for (int r = 0; r < 8; ++r) {
        const int row = bm + wm + mi * 16 + r + hi * 8;
        const int col = bn + wn + ni * 16 + cl;
        float vv = acc[mi][ni][r] + bias[col];
        const size_t idx = (size_t)row * N + col;
        if (EPI == EPI_H) {
          outH[idx] = (_Float16)vv;
        } else if (EPI == EPI_GELU) {
          float g = 0.5f * vv *
                    (1.f + tanhf(0.7978845608f * (vv + 0.044715f * vv * vv * vv)));
          outH[idx] = (_Float16)g;
        } else if (EPI == EPI_RES) {
          float o = resid[idx] + vv;
          outF[idx] = o;
          if (outH) outH[idx] = (_Float16)o;
        } else {  // EPI_GATERES
          float gate = sst[gateIdx * N + col] +
                       t6[(size_t)row * 6 * N + gateIdx * N + col];
          float o = resid[idx] + gate * vv;
          outF[idx] = o;
          if (outH) outH[idx] = (_Float16)o;
        }
      }
}

// ---------------------------------------------------------------------------
// Flash-style attention with WMMA. Layouts are (B, Seq, H, 128) f16.
// Block = (b, h, 128-query tile); 8 waves x 16 query rows; key blocks of 32.
// ---------------------------------------------------------------------------
__global__ __launch_bounds__(256) void attn_wmma_kernel(
    const _Float16* __restrict__ Q, const _Float16* __restrict__ Kp,
    const _Float16* __restrict__ Vp, _Float16* __restrict__ O,
    int H, int Sq, int Sk, float scale) {
  __shared__ __align__(16) _Float16 sK[32 * 128];     // key-major
  __shared__ __align__(16) _Float16 sV[128 * 32];     // d-major (transposed)
  __shared__ __align__(16) _Float16 sP[8][16 * 32];   // per-wave P scratch
  const int tid = threadIdx.x, lane = tid & 31, w = tid >> 5;
  const int qtiles = Sq >> 7;
  const int qt = blockIdx.x % qtiles;
  const int h  = (blockIdx.x / qtiles) % H;
  const int b  = blockIdx.x / (qtiles * H);
  const int qbase = qt * 128 + w * 16;
  const int lr = lane & 15, hi = lane >> 4;

  FragH qf[4];
  {
    const _Float16* gq = Q + ((size_t)(b * Sq + qbase + lr) * H + h) * 128;
    #pragma unroll
    for (int c = 0; c < 4; ++c) {
      qf[c].q[0] = *(const u32x4*)(gq + c * 32 + hi * 8);
      qf[c].q[1] = *(const u32x4*)(gq + c * 32 + 16 + hi * 8);
    }
  }
  float m[8], l[8];
  v8f o[8];
  #pragma unroll
  for (int r = 0; r < 8; ++r) { m[r] = -3e38f; l[r] = 0.f; }
  #pragma unroll
  for (int nt = 0; nt < 8; ++nt)
    #pragma unroll
    for (int r = 0; r < 8; ++r) o[nt][r] = 0.f;

  const int kr = tid >> 3, seg = (tid & 7) * 16;

  for (int kb = 0; kb < Sk; kb += 32) {
    __syncthreads();  // prior iteration's LDS consumers done
    {
      // K tile: async copy straight into LDS (key-major, contiguous)
      const _Float16* gk = Kp + ((size_t)(b * Sk + kb + kr) * H + h) * 128 + seg;
      async_cp_b128(gk,     &sK[kr * 128 + seg]);
      async_cp_b128(gk + 8, &sK[kr * 128 + seg + 8]);
      // V tile: transposed into LDS (needs 16-bit scatter, stays on VALU path)
      const _Float16* gv = Vp + ((size_t)(b * Sk + kb + kr) * H + h) * 128 + seg;
      union { u32x4 q[2]; _Float16 e[16]; } tv;
      tv.q[0] = ((const u32x4*)gv)[0];
      tv.q[1] = ((const u32x4*)gv)[1];
      #pragma unroll
      for (int j = 0; j < 16; ++j) sV[(seg + j) * 32 + kr] = tv.e[j];
    }
    wait_async<0>();
    __syncthreads();

    // scores S = Q K^T (16x32) as two 16x16 f32 fragments
    v8f s0, s1;
    #pragma unroll
    for (int r = 0; r < 8; ++r) { s0[r] = 0.f; s1[r] = 0.f; }
    #pragma unroll
    for (int c = 0; c < 4; ++c) {
      FragH fb0, fb1;
      load_frag_b(fb0, sK, 128, 0,  c * 32, lane);
      load_frag_b(fb1, sK, 128, 16, c * 32, lane);
      s0 = wmma_f16(qf[c], fb0, s0);
      s1 = wmma_f16(qf[c], fb1, s1);
    }

    // online softmax (rows live in half-waves -> width-16 shuffles)
    float alpha[8];
    #pragma unroll
    for (int r = 0; r < 8; ++r) {
      float a0 = s0[r] * scale, a1 = s1[r] * scale;
      float mloc = fmaxf(a0, a1);
      #pragma unroll
      for (int off = 8; off; off >>= 1) mloc = fmaxf(mloc, __shfl_xor(mloc, off, 16));
      float mn = fmaxf(m[r], mloc);
      alpha[r] = __expf(m[r] - mn);
      float p0 = __expf(a0 - mn), p1 = __expf(a1 - mn);
      float rsum = p0 + p1;
      #pragma unroll
      for (int off = 8; off; off >>= 1) rsum += __shfl_xor(rsum, off, 16);
      l[r] = l[r] * alpha[r] + rsum;
      m[r] = mn;
      s0[r] = p0; s1[r] = p1;
    }
    #pragma unroll
    for (int nt = 0; nt < 8; ++nt)
      #pragma unroll
      for (int r = 0; r < 8; ++r) o[nt][r] *= alpha[r];

    // D-fragment -> A-fragment relayout of P through per-wave LDS scratch
    _Float16* pw = sP[w];
    #pragma unroll
    for (int r = 0; r < 8; ++r) {
      const int row = r + hi * 8;
      pw[row * 32 + lr]      = (_Float16)s0[r];
      pw[row * 32 + 16 + lr] = (_Float16)s1[r];
    }
    __syncthreads();

    FragH fp;
    load_frag_a(fp, pw, 32, 0, lane);
    #pragma unroll
    for (int nt = 0; nt < 8; ++nt) {
      FragH fv;
      load_frag_b(fv, sV, 32, nt * 16, 0, lane);
      o[nt] = wmma_f16(fp, fv, o[nt]);   // O += P V
    }
  }

  #pragma unroll
  for (int r = 0; r < 8; ++r) {
    const float inv = 1.f / l[r];
    const int row = qbase + r + hi * 8;
    _Float16* go = O + ((size_t)(b * Sq + row) * H + h) * 128;
    #pragma unroll
    for (int nt = 0; nt < 8; ++nt)
      go[nt * 16 + lr] = (_Float16)(o[nt][r] * inv);
  }
}

// ---------------------------------------------------------------------------
// Host orchestration
// ---------------------------------------------------------------------------
extern "C" void kernel_launch(void* const* d_in, const int* in_sizes, int n_in,
                              void* d_out, int out_size, void* d_ws, size_t ws_size,
                              hipStream_t stream) {
  (void)in_sizes; (void)n_in; (void)out_size; (void)ws_size;
  const int C = 2048, H = 16, S = 1024, T = 256, B = 4, F = 8192;
  const int L = B * S, LY = B * T;
  const float scale = 0.08838834764831845f;  // 1/sqrt(128)

  const float* x     = (const float*)d_in[0];
  const float* y     = (const float*)d_in[1];
  const float* t6    = (const float*)d_in[2];
  const float* freqs = (const float*)d_in[3];
  const float* sst   = (const float*)d_in[4];
  const float* Wq_s = (const float*)d_in[5];   const float* bq_s = (const float*)d_in[6];
  const float* Wk_s = (const float*)d_in[7];   const float* bk_s = (const float*)d_in[8];
  const float* Wv_s = (const float*)d_in[9];   const float* bv_s = (const float*)d_in[10];
  const float* Wo_s = (const float*)d_in[11];  const float* bo_s = (const float*)d_in[12];
  const float* Wq_c = (const float*)d_in[13];  const float* bq_c = (const float*)d_in[14];
  const float* Wk_c = (const float*)d_in[15];  const float* bk_c = (const float*)d_in[16];
  const float* Wv_c = (const float*)d_in[17];  const float* bv_c = (const float*)d_in[18];
  const float* Wo_c = (const float*)d_in[19];  const float* bo_c = (const float*)d_in[20];
  const float* Wfc1 = (const float*)d_in[21];  const float* bfc1 = (const float*)d_in[22];
  const float* Wfc2 = (const float*)d_in[23];  const float* bfc2 = (const float*)d_in[24];
  float* out = (float*)d_out;

  char* wsb = (char*)d_ws;
  size_t off = 0;
  auto alloc = [&](size_t bytes) -> void* {
    void* p = wsb + off;
    off += (bytes + 255) & ~(size_t)255;
    return p;
  };
  _Float16* wq_s16 = (_Float16*)alloc((size_t)C * C * 2);
  _Float16* wk_s16 = (_Float16*)alloc((size_t)C * C * 2);
  _Float16* wv_s16 = (_Float16*)alloc((size_t)C * C * 2);
  _Float16* wo_s16 = (_Float16*)alloc((size_t)C * C * 2);
  _Float16* wq_c16 = (_Float16*)alloc((size_t)C * C * 2);
  _Float16* wk_c16 = (_Float16*)alloc((size_t)C * C * 2);
  _Float16* wv_c16 = (_Float16*)alloc((size_t)C * C * 2);
  _Float16* wo_c16 = (_Float16*)alloc((size_t)C * C * 2);
  _Float16* wfc1_16 = (_Float16*)alloc((size_t)F * C * 2);
  _Float16* wfc2_16 = (_Float16*)alloc((size_t)C * F * 2);
  _Float16* y16  = (_Float16*)alloc((size_t)LY * C * 2);
  _Float16* xm16 = (_Float16*)alloc((size_t)L * C * 2);
  _Float16* q16  = (_Float16*)alloc((size_t)L * C * 2);
  _Float16* k16  = (_Float16*)alloc((size_t)L * C * 2);
  _Float16* v16  = (_Float16*)alloc((size_t)L * C * 2);
  _Float16* a16  = (_Float16*)alloc((size_t)L * C * 2);
  _Float16* x1h  = (_Float16*)alloc((size_t)L * C * 2);
  _Float16* h16  = (_Float16*)alloc((size_t)L * F * 2);
  float* x1 = (float*)alloc((size_t)L * C * 4);
  float* x2 = (float*)alloc((size_t)L * C * 4);

  auto conv = [&](const float* src, _Float16* dst, int n) {
    f32_to_f16_kernel<<<(n / 4 + 255) / 256, 256, 0, stream>>>(src, dst, n);
  };
  conv(Wq_s, wq_s16, C * C);   conv(Wk_s, wk_s16, C * C);
  conv(Wv_s, wv_s16, C * C);   conv(Wo_s, wo_s16, C * C);
  conv(Wq_c, wq_c16, C * C);   conv(Wk_c, wk_c16, C * C);
  conv(Wv_c, wv_c16, C * C);   conv(Wo_c, wo_c16, C * C);
  conv(Wfc1, wfc1_16, F * C);  conv(Wfc2, wfc2_16, C * F);
  conv(y, y16, LY * C);

  const dim3 gCC(C / 128, L / 128);    // M=L,  N=C
  const dim3 gYC(C / 128, LY / 128);   // M=LY, N=C
  const dim3 gCF(F / 128, L / 128);    // M=L,  N=F

  // --- self-attention branch ---
  ln_mod_kernel<<<L, 256, 0, stream>>>(x, t6, sst, xm16, C, 1, 0);
  gemm_wmma_kernel<EPI_H><<<gCC, 256, 0, stream>>>(
      xm16, wq_s16, bq_s, q16, nullptr, nullptr, nullptr, nullptr, 0, L, C, C);
  gemm_wmma_kernel<EPI_H><<<gCC, 256, 0, stream>>>(
      xm16, wk_s16, bk_s, k16, nullptr, nullptr, nullptr, nullptr, 0, L, C, C);
  gemm_wmma_kernel<EPI_H><<<gCC, 256, 0, stream>>>(
      xm16, wv_s16, bv_s, v16, nullptr, nullptr, nullptr, nullptr, 0, L, C, C);
  {
    const int nPairs = L * H * 64;
    rope_kernel<<<(nPairs + 255) / 256, 256, 0, stream>>>(q16, freqs, nPairs, H, 64, S);
    rope_kernel<<<(nPairs + 255) / 256, 256, 0, stream>>>(k16, freqs, nPairs, H, 64, S);
  }
  attn_wmma_kernel<<<B * H * (S / 128), 256, 0, stream>>>(
      q16, k16, v16, a16, H, S, S, scale);
  gemm_wmma_kernel<EPI_GATERES><<<gCC, 256, 0, stream>>>(
      a16, wo_s16, bo_s, x1h, x1, x, t6, sst, 2, L, C, C);

  // --- cross-attention branch ---
  gemm_wmma_kernel<EPI_H><<<gCC, 256, 0, stream>>>(
      x1h, wq_c16, bq_c, q16, nullptr, nullptr, nullptr, nullptr, 0, L, C, C);
  gemm_wmma_kernel<EPI_H><<<gYC, 256, 0, stream>>>(
      y16, wk_c16, bk_c, k16, nullptr, nullptr, nullptr, nullptr, 0, LY, C, C);
  gemm_wmma_kernel<EPI_H><<<gYC, 256, 0, stream>>>(
      y16, wv_c16, bv_c, v16, nullptr, nullptr, nullptr, nullptr, 0, LY, C, C);
  attn_wmma_kernel<<<B * H * (S / 128), 256, 0, stream>>>(
      q16, k16, v16, a16, H, S, T, scale);
  gemm_wmma_kernel<EPI_RES><<<gCC, 256, 0, stream>>>(
      a16, wo_c16, bo_c, nullptr, x2, x1, nullptr, nullptr, 0, L, C, C);

  // --- MLP branch ---
  ln_mod_kernel<<<L, 256, 0, stream>>>(x2, t6, sst, xm16, C, 4, 3);
  gemm_wmma_kernel<EPI_GELU><<<gCF, 256, 0, stream>>>(
      xm16, wfc1_16, bfc1, h16, nullptr, nullptr, nullptr, nullptr, 0, L, F, C);
  gemm_wmma_kernel<EPI_GATERES><<<gCC, 256, 0, stream>>>(
      h16, wfc2_16, bfc2, nullptr, out, x2, t6, sst, 5, L, C, F);
}
